// ShallowTopK_8486855377011
// MI455X (gfx1250) — matrix operand order repaired and checked by reading, not compiled, not run on hardware
//
#include <hip/hip_runtime.h>
#include <stdint.h>

// ---------------- problem dims ----------------
#define D_MODEL 768
#define D_FEAT  24576
#define NTOK    8192        // B*S = 4*2048
#define KTOP    32
#define M_TILE  16          // tokens per workgroup
#define NF_ITER 128         // features per outer iteration (8 waves x 16)
#define NITER   (D_FEAT / NF_ITER)   // 192
#define AUX_COEFF 0.03125f

// ---------------- LDS layout (bytes), encode kernel ----------------
#define XH_OFF     0        // bf16 hi  [16][768] padded rows
#define XH_STRIDE  1552     // 768*2 bytes padded to 1552 (stride%64dw==4 -> conflict free)
#define XL_OFF     24832    // bf16 lo
#define PRE_OFF    49664    // f32 staged tile [16][132]
#define PRE_STRIDE 132      // dwords
#define TOPV_OFF   58112    // f32 [16][32]
#define TOPI_OFF   60160    // i32 [16][32]
#define SMEM_BYTES 62208

typedef __attribute__((ext_vector_type(8)))  float        v8f;
typedef __attribute__((ext_vector_type(16))) __bf16       v16bf;
typedef __attribute__((ext_vector_type(8)))  __bf16       v8bf;
typedef __attribute__((ext_vector_type(4)))  unsigned int v4u;
typedef __attribute__((ext_vector_type(4)))  int          v4i;
typedef __attribute__((ext_vector_type(8)))  int          v8i;

union BF16x16 { v16bf v; v8bf h[2]; __bf16 e[16]; };
union BF16x8  { v8bf  v; __bf16 e[8]; };

// ============================================================
// Kernel 0b: one-time transpose + bf16 hi/lo split of W_enc
//   W_enc[768][24576] f32  ->  Whi/Wlo[24576][768] bf16 (K-major rows)
// This makes the encoder's WMMA B-operand two contiguous b128 loads
// per matrix half with zero in-loop conversion VALU.
// ============================================================
#define TKT 32   // K rows per tile
#define TFT 64   // features per tile
__global__ __launch_bounds__(256)
void sae_split_wenc(const float* __restrict__ W_enc,
                    __bf16* __restrict__ Whi,
                    __bf16* __restrict__ Wlo)
{
    __shared__ float tile[TKT][TFT + 1];   // +1 pad: conflict-free transpose
    const int t  = threadIdx.x;
    const int k0 = blockIdx.x * TKT;       // 0..767
    const int f0 = blockIdx.y * TFT;       // 0..24575
    // load 32x64 f32 tile, coalesced along features
    {
        const int kr = t >> 6;             // 0..3
        const int fc = t & 63;
        #pragma unroll
        for (int i = 0; i < 8; ++i) {
            int k = kr + i * 4;
            tile[k][fc] = W_enc[(size_t)(k0 + k) * D_FEAT + (f0 + fc)];
        }
    }
    __syncthreads();
    // write 64 feature-rows x 32 K, coalesced along K (16B vector stores)
    {
        const int fr  = t >> 2;            // 0..63
        const int kc0 = (t & 3) * 8;       // 0,8,16,24
        BF16x8 hb, lb;
        #pragma unroll
        for (int j = 0; j < 8; ++j) {
            float  v = tile[kc0 + j][fr];
            __bf16 h = (__bf16)v;
            hb.e[j] = h;
            lb.e[j] = (__bf16)(v - (float)h);
        }
        size_t base = (size_t)(f0 + fr) * D_MODEL + (size_t)(k0 + kc0);
        *(v8bf*)(Whi + base) = hb.v;
        *(v8bf*)(Wlo + base) = lb.v;
    }
}

// ============================================================
// Kernel 1: fused encoder GEMM (bf16-split WMMA) + per-token top-32
// ============================================================
__global__ __launch_bounds__(256)
void sae_encode_topk(const float* __restrict__ x,
                     const __bf16* __restrict__ Whi,
                     const __bf16* __restrict__ Wlo,
                     const float* __restrict__ b_enc,
                     float* __restrict__ wsv,
                     int*   __restrict__ wsi)
{
    __shared__ __align__(16) unsigned char smem[SMEM_BYTES];
    const int t    = threadIdx.x;
    const int wv   = t >> 5;          // wave id 0..7
    const int lane = t & 31;
    const int hl   = lane >> 4;       // lane half (0/1)
    const int fn   = lane & 15;
    const int tok0 = blockIdx.x * M_TILE;

    // ---- TDM: DMA the 16x768 fp32 x-tile into LDS (bytes [0,49152)) ----
    if (t < 32) {
        uint64_t ga = (uint64_t)(uintptr_t)(x + (size_t)tok0 * D_MODEL);
        v4u g0;
        g0.x = 1u;                                     // count=1 user descriptor
        g0.y = (unsigned)XH_OFF;                       // lds_addr (byte)
        g0.z = (unsigned)(ga & 0xFFFFFFFFu);           // global_addr[31:0]
        g0.w = (unsigned)((ga >> 32) & 0x1FFFFFFu) | 0x80000000u; // addr[56:32] | type=2
        v8i g1;
        g1[0] = 0x00020000;                            // data_size = 4B
        g1[1] = (int)(((unsigned)D_MODEL & 0xFFFFu) << 16); // tensor_dim0 lo16
        g1[2] = (int)(((unsigned)NTOK & 0xFFFFu) << 16);    // tensor_dim1 lo16
        g1[3] = (int)((unsigned)D_MODEL << 16);        // tile_dim0 = 768
        g1[4] = M_TILE;                                // tile_dim1 = 16 rows
        g1[5] = D_MODEL;                               // tensor_dim0_stride
        g1[6] = 0;
        g1[7] = 0;
        v4i gz = {0, 0, 0, 0};
#if defined(__clang_major__) && (__clang_major__ >= 23)
        v8i gz8 = {0, 0, 0, 0, 0, 0, 0, 0};
        __builtin_amdgcn_tensor_load_to_lds(g0, g1, gz, gz, gz8, 0);
#else
        __builtin_amdgcn_tensor_load_to_lds(g0, g1, gz, gz, 0);
#endif
        __builtin_amdgcn_s_wait_tensorcnt(0);
    }
    __syncthreads();

    // ---- split x tile into bf16 hi/lo (in-place over the staged f32 region,
    //      via register staging: 48 elements per thread) ----
    {
        const float* xsf = (const float*)(smem);  // staged f32 tile, 12288 floats
        float tmp[48];
        #pragma unroll
        for (int i = 0; i < 48; ++i) tmp[i] = xsf[t + i * 256];
        __syncthreads();
        #pragma unroll
        for (int i = 0; i < 48; ++i) {
            int e   = t + i * 256;
            int row = e / D_MODEL;
            int col = e - row * D_MODEL;
            float  v = tmp[i];
            __bf16 h = (__bf16)v;
            float  r = v - (float)h;
            *(__bf16*)(smem + XH_OFF + row * XH_STRIDE + col * 2) = h;
            *(__bf16*)(smem + XL_OFF + row * XH_STRIDE + col * 2) = (__bf16)r;
        }
    }
    // init top-k lists (region disjoint from x tile)
    float* topv = (float*)(smem + TOPV_OFF);
    int*   topi = (int*)(smem + TOPI_OFF);
    for (int e = t; e < M_TILE * KTOP; e += 256) { topv[e] = -1.0f; topi[e] = 0; }
    __syncthreads();

    float* pre  = (float*)(smem + PRE_OFF);
    float  curmin = -1.0f;     // register-cached min of this token's top list (t<16 only)
    int    cms    = 0;

    const unsigned abase = (unsigned)(fn * XH_STRIDE + hl * 16); // A byte base per lane

    for (int itr = 0; itr < NITER; ++itr) {
        const int   f0   = itr * NF_ITER + wv * 16;
        const float bias = b_enc[f0 + fn];
        v8f c = {bias, bias, bias, bias, bias, bias, bias, bias};
        // pre-split, K-major W_enc rows for this lane's feature column
        const __bf16* whrow = Whi + (size_t)(f0 + fn) * D_MODEL;
        const __bf16* wlrow = Wlo + (size_t)(f0 + fn) * D_MODEL;

        #pragma unroll 2
        for (int kb = 0; kb < D_MODEL; kb += 32) {
            // A operand: 16x32 bf16, lane<16 holds K {kb..kb+7, kb+16..kb+23}
            BF16x16 ah, al;
            ah.h[0] = *(const v8bf*)(smem + XH_OFF + abase + kb * 2);
            ah.h[1] = *(const v8bf*)(smem + XH_OFF + abase + kb * 2 + 32);
            al.h[0] = *(const v8bf*)(smem + XL_OFF + abase + kb * 2);
            al.h[1] = *(const v8bf*)(smem + XL_OFF + abase + kb * 2 + 32);

            // B operand: 32x16 bf16; lane<16 K {kb..kb+7, kb+16..kb+23}, lane>=16 K {kb+8..kb+15, kb+24..kb+31}
            // contiguous b128 loads, no conversion VALU
            const int kbase = kb + hl * 8;
            BF16x16 bh, bl;
            bh.h[0] = *(const v8bf*)(whrow + kbase);
            bh.h[1] = *(const v8bf*)(whrow + kbase + 16);
            bl.h[0] = *(const v8bf*)(wlrow + kbase);
            bl.h[1] = *(const v8bf*)(wlrow + kbase + 16);
            if (kb + 64 < D_MODEL)
                __builtin_prefetch(whrow + kbase + 64, 0, 1);

            // split-precision f32 ~= hi*hi + hi*lo + lo*hi, f32 accumulate
            c = __builtin_amdgcn_wmma_f32_16x16x32_bf16(false, ah.v, false, bh.v,
                                                        (short)0, c, false, false);
            c = __builtin_amdgcn_wmma_f32_16x16x32_bf16(false, ah.v, false, bl.v,
                                                        (short)0, c, false, false);
            c = __builtin_amdgcn_wmma_f32_16x16x32_bf16(false, al.v, false, bh.v,
                                                        (short)0, c, false, false);
        }
        // fused ReLU
        #pragma unroll
        for (int r = 0; r < 8; ++r) c[r] = c[r] > 0.0f ? c[r] : 0.0f;

        __syncthreads();   // previous merge finished reading `pre`
        #pragma unroll
        for (int r = 0; r < 8; ++r)
            pre[(r + 8 * hl) * PRE_STRIDE + wv * 16 + fn] = c[r];
        __syncthreads();

        // top-32 merge: one thread per token, insertions are rare after warmup
        if (t < M_TILE) {
            float* tv = topv + t * KTOP;
            int*   ti = topi + t * KTOP;
            for (int j = 0; j < NF_ITER; ++j) {
                float v = pre[t * PRE_STRIDE + j];
                if (v > curmin) {
                    tv[cms] = v;
                    ti[cms] = itr * NF_ITER + j;
                    curmin = tv[0]; cms = 0;
                    #pragma unroll 8
                    for (int s = 1; s < KTOP; ++s) {
                        float q = tv[s];
                        if (q < curmin) { curmin = q; cms = s; }
                    }
                }
            }
        }
    }
    __syncthreads();

    // write per-token top-32 (value,index) to workspace
    for (int e = t; e < M_TILE * KTOP; e += 256) {
        int tok = e >> 5, s = e & 31;
        size_t g = ((size_t)tok0 + tok) * KTOP + s;
        float  v = topv[e];
        wsv[g] = v > 0.0f ? v : 0.0f;   // clamp -1 sentinels
        wsi[g] = topi[e];
    }
}

// ============================================================
// Kernel 2: sparse decode + aux decode + fused L2/aux loss reduction
// ============================================================
__global__ __launch_bounds__(256)
void sae_decode_loss(const float* __restrict__ x,
                     const float* __restrict__ W_dec,
                     const float* __restrict__ b_dec,
                     const float* __restrict__ n_inactive,
                     const float* __restrict__ wsv,
                     const int*   __restrict__ wsi,
                     float* __restrict__ recon,
                     float* __restrict__ sums)
{
    __shared__ float sv[KTOP];
    __shared__ float sm[KTOP];
    __shared__ int   si[KTOP];
    __shared__ float red[16];
    const int n = blockIdx.x;
    const int t = threadIdx.x;
    if (t < KTOP) {
        float v = wsv[(size_t)n * KTOP + t];
        int   i = wsi[(size_t)n * KTOP + t];
        sv[t] = v;
        si[t] = i;
        sm[t] = (n_inactive[i] > 100.0f) ? 1.0f : 0.0f;   // dead-feature mask
    }
    __syncthreads();

    float l2 = 0.0f, aux = 0.0f;
    #pragma unroll
    for (int rep = 0; rep < 3; ++rep) {
        int d = t + rep * 256;
        float acc  = b_dec[d];
        float accA = acc;
        #pragma unroll 8
        for (int i = 0; i < KTOP; ++i) {
            float w = W_dec[(size_t)si[i] * D_MODEL + d];
            acc  += sv[i] * w;
            accA += sv[i] * sm[i] * w;
        }
        float xd   = x[(size_t)n * D_MODEL + d];
        float diff = acc - xd;             // recon - input
        l2 += diff * diff;
        float ad = accA - (xd - acc);      // recon_aux - residual
        aux += ad * ad;
        recon[(size_t)n * D_MODEL + d] = acc;
    }

    // wave32 reduction then cross-wave + global atomic
    #pragma unroll
    for (int m = 16; m >= 1; m >>= 1) {
        l2  += __shfl_xor(l2,  m, 32);
        aux += __shfl_xor(aux, m, 32);
    }
    int wvid = t >> 5;
    if ((t & 31) == 0) { red[wvid] = l2; red[8 + wvid] = aux; }
    __syncthreads();
    if (t == 0) {
        float L = 0.0f, A = 0.0f;
        #pragma unroll
        for (int i = 0; i < 8; ++i) { L += red[i]; A += red[8 + i]; }
        atomicAdd(&sums[0], L);
        atomicAdd(&sums[1], A);
    }
}

// ============================================================
// init / finalize
// ============================================================
__global__ void sae_init(float* __restrict__ sums)
{
    if (threadIdx.x < 2) sums[threadIdx.x] = 0.0f;
}

__global__ void sae_finalize(const float* __restrict__ sums, float* __restrict__ out3)
{
    const float inv = 1.0f / (float)((size_t)NTOK * D_MODEL);
    float l2  = sums[0] * inv;
    float aux = sums[1] * inv;
    out3[0] = l2 + AUX_COEFF * aux;   // loss
    out3[1] = l2;                     // l2_loss
    out3[2] = aux;                    // aux_loss
}

// ============================================================
extern "C" void kernel_launch(void* const* d_in, const int* in_sizes, int n_in,
                              void* d_out, int out_size, void* d_ws, size_t ws_size,
                              hipStream_t stream)
{
    (void)in_sizes; (void)n_in; (void)out_size; (void)ws_size;
    const float* x       = (const float*)d_in[0];
    const float* W_enc   = (const float*)d_in[1];
    const float* b_enc   = (const float*)d_in[2];
    const float* W_dec   = (const float*)d_in[3];
    const float* b_dec   = (const float*)d_in[4];
    const float* n_inact = (const float*)d_in[5];

    float* recon = (float*)d_out;
    float* out3  = recon + (size_t)NTOK * D_MODEL;

    // workspace layout:
    //   [0, 1MB)        top-32 values
    //   [1MB, 2MB)      top-32 indices
    //   [2MB, 2MB+8)    loss accumulators
    //   [2MB+256, ...)  Whi bf16 [24576][768], then Wlo bf16 [24576][768]
    float* wsv  = (float*)d_ws;
    int*   wsi  = (int*)((char*)d_ws + (size_t)NTOK * KTOP * 4);
    float* sums = (float*)((char*)d_ws + (size_t)2 * NTOK * KTOP * 4);
    __bf16* Whi = (__bf16*)((char*)d_ws + (size_t)2 * NTOK * KTOP * 4 + 256);
    __bf16* Wlo = Whi + (size_t)D_FEAT * D_MODEL;

    sae_init<<<1, 32, 0, stream>>>(sums);
    sae_split_wenc<<<dim3(D_MODEL / TKT, D_FEAT / TFT), 256, 0, stream>>>(W_enc, Whi, Wlo);
    sae_encode_topk<<<NTOK / M_TILE, 256, 0, stream>>>(x, Whi, Wlo, b_enc, wsv, wsi);
    sae_decode_loss<<<NTOK, 256, 0, stream>>>(x, W_dec, b_dec, n_inact, wsv, wsi, recon, sums);
    sae_finalize<<<1, 1, 0, stream>>>(sums, out3);
}
